// LimitedRelPositionMultiHeadedAttention_49615462204126
// MI455X (gfx1250) — matrix operand order repaired and checked by reading, not compiled, not run on hardware
//
#include <hip/hip_runtime.h>
#include <hip/hip_bf16.h>

typedef __attribute__((ext_vector_type(16))) __bf16 bf16x16;
typedef __attribute__((ext_vector_type(8)))  __bf16 bf16x8;
typedef __attribute__((ext_vector_type(4)))  __bf16 bf16x4;
typedef __attribute__((ext_vector_type(8)))  float  f32x8;

static constexpr int T_SEQ = 1024;
static constexpr int FDIM  = 512;
static constexpr int HEADS = 8;
static constexpr int DKV   = 64;
static constexpr int WBAND = 128;

union FragU { bf16x16 v; bf16x8 h[2]; };

__device__ inline bf16x16 load_frag2(const __bf16* p0, const __bf16* p1) {
  FragU u;
  u.h[0] = *(const bf16x8*)p0;
  u.h[1] = *(const bf16x8*)p1;
  return u.v;
}

__device__ inline f32x8 wmma_bf16(bf16x16 a, bf16x16 b, f32x8 c) {
  return __builtin_amdgcn_wmma_f32_16x16x32_bf16(
      false, a, false, b, (short)0, c, false, false);
}

// ---------------------------------------------------------------------------
// Tiled bf16 GEMM:  C(MxN) = A(Mx512) * W(512xN) (+bias), N = 512 fixed.
// Block tile 128x128, K-step 32. 8 waves as 4(M) x 2(N), each wave 32x64.
// EPI 0: bf16 row-major out (+bias)                      [K proj, P proj]
// EPI 1: dual bf16 row-major out (+bias+u / +bias+v)     [Q proj]
// EPI 2: bf16 transposed (B,H,DK,T) out (+bias)          [V proj]
// EPI 3: f32 row-major out (+bias)                       [O proj]
// ---------------------------------------------------------------------------
template<int EPI, bool ABF16>
__global__ void __launch_bounds__(256)
gemm512(const void* __restrict__ Aptr, const float* __restrict__ Wmat,
        const float* __restrict__ bias, const float* __restrict__ vecU,
        const float* __restrict__ vecV,
        __bf16* __restrict__ out0, __bf16* __restrict__ out1,
        float* __restrict__ outF, int M)
{
  __shared__ __bf16 aLds[128][32];   // [m][k]
  __shared__ __bf16 wLds[128][32];   // [n][k]  (W staged transposed)

  const int tid  = threadIdx.x;
  const int lane = tid & 31;
  const int l15  = lane & 15;
  const int half = lane >> 4;
  const int wv   = tid >> 5;
  const int wm   = wv & 3;      // 0..3 along M
  const int wn   = wv >> 2;     // 0..1 along N
  const int m0   = blockIdx.y * 128;
  const int n0   = blockIdx.x * 128;

  const float*  Af = (const float*)Aptr;
  const __bf16* Ab = (const __bf16*)Aptr;

  f32x8 acc[2][4];
  #pragma unroll
  for (int i = 0; i < 2; ++i)
    #pragma unroll
    for (int j = 0; j < 4; ++j) acc[i][j] = (f32x8){};

  for (int k0 = 0; k0 < 512; k0 += 32) {
    // stage A tile (128x32): b128 global loads, b64 LDS stores
    #pragma unroll
    for (int it = 0; it < 4; ++it) {
      int f4 = it * 256 + tid;            // float4 index within 128x32
      int r  = f4 >> 3, c4 = (f4 & 7) * 4;
      int gm = m0 + r;
      bf16x4 o = {};
      if (ABF16) {
        if (gm < M) o = *(const bf16x4*)(Ab + gm * 512 + k0 + c4);
      } else {
        float4 x = {0.0f, 0.0f, 0.0f, 0.0f};
        if (gm < M) x = *(const float4*)(Af + gm * 512 + k0 + c4);
        o[0] = (__bf16)x.x; o[1] = (__bf16)x.y;
        o[2] = (__bf16)x.z; o[3] = (__bf16)x.w;
      }
      *(bf16x4*)(&aLds[r][c4]) = o;
    }
    // stage W tile (32x128): b128 global loads, transposed b16 LDS stores
    #pragma unroll
    for (int it = 0; it < 4; ++it) {
      int f4  = it * 256 + tid;
      int kk2 = f4 >> 5, n4 = (f4 & 31) * 4;
      float4 x = *(const float4*)(Wmat + (k0 + kk2) * 512 + n0 + n4);
      wLds[n4 + 0][kk2] = (__bf16)x.x;
      wLds[n4 + 1][kk2] = (__bf16)x.y;
      wLds[n4 + 2][kk2] = (__bf16)x.z;
      wLds[n4 + 3][kk2] = (__bf16)x.w;
    }
    // prefetch next K-step (global_prefetch_b8)
    if (k0 + 32 < 512) {
      int pr = tid >> 1;
      int gm = m0 + pr;
      if (gm < M) {
        if (ABF16) __builtin_prefetch(Ab + gm * 512 + k0 + 32 + (tid & 1) * 16, 0, 1);
        else       __builtin_prefetch(Af + gm * 512 + k0 + 32 + (tid & 1) * 16, 0, 1);
      }
      __builtin_prefetch(Wmat + (k0 + 32 + (tid >> 3)) * 512 + n0 + (tid & 7) * 16, 0, 1);
    }
    __syncthreads();

    bf16x16 afr[2], bfr[4];
    #pragma unroll
    for (int am = 0; am < 2; ++am) {
      const __bf16* row = &aLds[wm * 32 + am * 16 + l15][0];
      afr[am] = load_frag2(row + 16 * half, row + 16 * half + 8);
    }
    #pragma unroll
    for (int bn = 0; bn < 4; ++bn) {
      const __bf16* row = &wLds[wn * 64 + bn * 16 + l15][0];
      bfr[bn] = load_frag2(row + 16 * half, row + 16 * half + 8);
    }
    #pragma unroll
    for (int am = 0; am < 2; ++am)
      #pragma unroll
      for (int bn = 0; bn < 4; ++bn)
        acc[am][bn] = wmma_bf16(afr[am], bfr[bn], acc[am][bn]);
    __syncthreads();
  }

  // epilogue
  #pragma unroll
  for (int am = 0; am < 2; ++am) {
    int mBase = m0 + wm * 32 + am * 16 + 8 * half;
    #pragma unroll
    for (int bn = 0; bn < 4; ++bn) {
      int n = n0 + wn * 64 + bn * 16 + l15;
      float bb = bias ? bias[n] : 0.0f;
      float uu = (EPI == 1) ? vecU[n] : 0.0f;
      float vb = (EPI == 1) ? vecV[n] : 0.0f;
      #pragma unroll
      for (int g = 0; g < 8; ++g) {
        int m = mBase + g;
        if (m >= M) continue;
        float c = acc[am][bn][g] + bb;
        if (EPI == 0) {
          out0[m * 512 + n] = (__bf16)c;
        } else if (EPI == 1) {
          out0[m * 512 + n] = (__bf16)(c + uu);
          out1[m * 512 + n] = (__bf16)(c + vb);
        } else if (EPI == 2) {
          int b_ = m >> 10, t = m & 1023;
          int hh = n >> 6,  d = n & 63;
          out0[(((b_ * HEADS + hh) * DKV + d) << 10) + t] = (__bf16)c;
        } else {
          outF[m * 512 + n] = c;
        }
      }
    }
  }
}

// ---------------------------------------------------------------------------
// Banded rel-pos attention. 1 wave per 16-query tile, 4 waves per block.
// Phase 1: bd_raw[16][288] = q_v . p over fixed r-window [880,1168) -> LDS
// Phase 2: 17 key tiles: AC via WMMA, fuse rel-shifted BD from LDS, online
//          softmax (shfl_xor row reductions), attn.V via WMMA pairs (K=32).
// WMMA pairs use independent accumulators to avoid the 5-NOP RAW hazard.
// ---------------------------------------------------------------------------
__global__ void __launch_bounds__(128)
attn_band(const __bf16* __restrict__ qu, const __bf16* __restrict__ qv,
          const __bf16* __restrict__ kk, const __bf16* __restrict__ vt,
          const __bf16* __restrict__ pp, __bf16* __restrict__ xo)
{
  __shared__ __bf16 bdLds[4][16][288];
  __shared__ __bf16 probs[4][16][32];

  const int tid  = threadIdx.x;
  const int lane = tid & 31;
  const int l15  = lane & 15;
  const int half = lane >> 4;
  const int wv   = tid >> 5;

  int qt = blockIdx.x * 4 + wv;
  const int i0 = (qt & 63) << 4;
  qt >>= 6;
  const int h = qt & 7;
  const int b = qt >> 3;

  // q_u / q_v A-fragments (16 queries x 64 dk, two K-steps of 32)
  const int qrow = (b * T_SEQ + i0 + l15) * FDIM + h * DKV;
  bf16x16 aU[2], aV[2];
  #pragma unroll
  for (int s = 0; s < 2; ++s) {
    int d0 = 32 * s + 16 * half;
    aU[s] = load_frag2(qu + qrow + d0, qu + qrow + d0 + 8);
    aV[s] = load_frag2(qv + qrow + d0, qv + qrow + d0 + 8);
  }

  // Phase 1: bd_raw for r in [880, 1168)   (r = j - i + T - 1)
  #pragma unroll 1
  for (int rt = 0; rt < 18; ++rt) {
    const __bf16* prow = pp + (880 + rt * 16 + l15) * FDIM + h * DKV;
    bf16x16 b0 = load_frag2(prow + 16 * half, prow + 16 * half + 8);
    bf16x16 b1 = load_frag2(prow + 32 + 16 * half, prow + 32 + 16 * half + 8);
    f32x8 c0 = (f32x8){}, c1 = (f32x8){};
    c0 = wmma_bf16(aV[0], b0, c0);
    c1 = wmma_bf16(aV[1], b1, c1);
    f32x8 c = c0 + c1;
    #pragma unroll
    for (int g = 0; g < 8; ++g)
      bdLds[wv][g + 8 * half][rt * 16 + l15] = (__bf16)c[g];
  }

  float rmax[8], rsum[8];
  f32x8 oacc[4];
  #pragma unroll
  for (int g = 0; g < 8; ++g) { rmax[g] = -1e30f; rsum[g] = 0.0f; }
  #pragma unroll
  for (int nt = 0; nt < 4; ++nt) oacc[nt] = (f32x8){};

  // Phase 2: 17 key tiles over j in [i0-128, i0+144)
  #pragma unroll 1
  for (int kt = 0; kt <= 16; ++kt) {
    int j0 = i0 - 128 + kt * 16;
    int jj = j0 + l15;
    int jc = jj < 0 ? 0 : (jj > T_SEQ - 1 ? T_SEQ - 1 : jj);
    // prefetch next key tile's row
    int jn = jj + 16;
    jn = jn < 0 ? 0 : (jn > T_SEQ - 1 ? T_SEQ - 1 : jn);
    __builtin_prefetch(kk + (b * T_SEQ + jn) * FDIM + h * DKV, 0, 1);

    const __bf16* krow = kk + (b * T_SEQ + jc) * FDIM + h * DKV;
    bf16x16 b0 = load_frag2(krow + 16 * half, krow + 16 * half + 8);
    bf16x16 b1 = load_frag2(krow + 32 + 16 * half, krow + 32 + 16 * half + 8);
    f32x8 c0 = (f32x8){}, c1 = (f32x8){};
    c0 = wmma_bf16(aU[0], b0, c0);
    c1 = wmma_bf16(aU[1], b1, c1);
    f32x8 c = c0 + c1;

    bool vj = (jj >= 0) && (jj < T_SEQ);
    #pragma unroll
    for (int g = 0; g < 8; ++g) {
      int m = g + 8 * half;
      int dd = jj - (i0 + m);
      bool valid = vj && (dd <= WBAND) && (dd >= -WBAND);
      float s_ = valid
          ? (c[g] + (float)bdLds[wv][m][kt * 16 + 15 + l15 - m]) * 0.125f
          : -1e30f;
      float tm = s_;
      tm = fmaxf(tm, __shfl_xor(tm, 1));
      tm = fmaxf(tm, __shfl_xor(tm, 2));
      tm = fmaxf(tm, __shfl_xor(tm, 4));
      tm = fmaxf(tm, __shfl_xor(tm, 8));
      float nm = fmaxf(rmax[g], tm);
      float sc = __expf(rmax[g] - nm);
      float p  = valid ? __expf(s_ - nm) : 0.0f;
      float ps = p;
      ps += __shfl_xor(ps, 1);
      ps += __shfl_xor(ps, 2);
      ps += __shfl_xor(ps, 4);
      ps += __shfl_xor(ps, 8);
      rsum[g] = rsum[g] * sc + ps;
      rmax[g] = nm;
      #pragma unroll
      for (int nt = 0; nt < 4; ++nt) oacc[nt][g] *= sc;
      probs[wv][m][(kt & 1) * 16 + l15] = (__bf16)p;
    }
    if (kt == 16) {
      #pragma unroll
      for (int g = 0; g < 8; ++g)
        probs[wv][g + 8 * half][16 + l15] = (__bf16)0.0f;
    }
    if ((kt & 1) || (kt == 16)) {          // attn . V over a 32-key pair
      int j0p = (kt & 1) ? (j0 - 16) : j0;
      const __bf16* prow = &probs[wv][l15][0];
      bf16x16 ap = load_frag2(prow + 16 * half, prow + 16 * half + 8);
      int js = j0p + 16 * half;
      js = js < 0 ? 0 : (js > T_SEQ - 16 ? T_SEQ - 16 : js);
      #pragma unroll
      for (int nt = 0; nt < 4; ++nt) {
        int d = nt * 16 + l15;
        const __bf16* vrow = vt + (((b * HEADS + h) * DKV + d) * T_SEQ) + js;
        bf16x16 bv = load_frag2(vrow, vrow + 8);
        oacc[nt] = wmma_bf16(ap, bv, oacc[nt]);
      }
    }
  }

  // normalize + store x as bf16 (B,T,F) for the O projection
  #pragma unroll
  for (int g = 0; g < 8; ++g) {
    int m = g + 8 * half;
    float inv = (rsum[g] > 0.0f) ? (1.0f / rsum[g]) : 0.0f;
    #pragma unroll
    for (int nt = 0; nt < 4; ++nt) {
      int d = nt * 16 + l15;
      xo[(b * T_SEQ + i0 + m) * FDIM + h * DKV + d] = (__bf16)(oacc[nt][g] * inv);
    }
  }
}

// ---------------------------------------------------------------------------
extern "C" void kernel_launch(void* const* d_in, const int* in_sizes, int n_in,
                              void* d_out, int out_size, void* d_ws, size_t ws_size,
                              hipStream_t stream) {
  const float* query   = (const float*)d_in[0];
  const float* key     = (const float*)d_in[1];
  const float* value   = (const float*)d_in[2];
  const float* pos_emb = (const float*)d_in[3];
  const float* Wq = (const float*)d_in[4];
  const float* bq = (const float*)d_in[5];
  const float* Wk = (const float*)d_in[6];
  const float* bk = (const float*)d_in[7];
  const float* Wv = (const float*)d_in[8];
  const float* bv = (const float*)d_in[9];
  const float* Wp = (const float*)d_in[10];
  const float* Wo = (const float*)d_in[11];
  const float* bo = (const float*)d_in[12];
  const float* pbu = (const float*)d_in[13];  // (H,DK) flat = 512 n-vector
  const float* pbv = (const float*)d_in[14];
  // d_in[15] = mask (unused; band recomputed analytically)

  const int B = in_sizes[0] / (T_SEQ * FDIM);
  const int M = B * T_SEQ;
  const int P = 2 * T_SEQ - 1;       // 2047

  // workspace layout (bf16 segments)
  size_t seg = (size_t)M * FDIM * sizeof(__bf16);
  char* ws = (char*)d_ws;
  __bf16* qu = (__bf16*)(ws);
  __bf16* qv = (__bf16*)(ws + seg);
  __bf16* kw = (__bf16*)(ws + 2 * seg);
  __bf16* vt = (__bf16*)(ws + 3 * seg);
  __bf16* pw = (__bf16*)(ws + 4 * seg);
  __bf16* xw = (__bf16*)(ws + 4 * seg + (size_t)2048 * FDIM * sizeof(__bf16));
  float*  outF = (float*)d_out;

  dim3 blk(256);
  dim3 grdM(4, (M + 127) / 128);
  dim3 grdP(4, (P + 127) / 128);

  // Q projection -> q_u, q_v (bias + pos_bias_u/v folded in)
  gemm512<1, false><<<grdM, blk, 0, stream>>>(query, Wq, bq, pbu, pbv,
                                              qu, qv, nullptr, M);
  // K projection -> row-major bf16
  gemm512<0, false><<<grdM, blk, 0, stream>>>(key, Wk, bk, nullptr, nullptr,
                                              kw, nullptr, nullptr, M);
  // V projection -> transposed (B,H,DK,T) bf16
  gemm512<2, false><<<grdM, blk, 0, stream>>>(value, Wv, bv, nullptr, nullptr,
                                              vt, nullptr, nullptr, M);
  // P projection (no bias) -> row-major bf16
  gemm512<0, false><<<grdP, blk, 0, stream>>>(pos_emb, Wp, nullptr, nullptr,
                                              nullptr, pw, nullptr, nullptr, P);
  // banded rel-pos attention
  dim3 ablk(128);
  dim3 agrd((B * HEADS * (T_SEQ / 16)) / 4);
  attn_band<<<agrd, ablk, 0, stream>>>(qu, qv, kw, vt, pw, xw);
  // O projection -> f32 out
  gemm512<3, true><<<grdM, blk, 0, stream>>>(xw, Wo, bo, nullptr, nullptr,
                                             nullptr, nullptr, outF, M);
  (void)n_in; (void)out_size; (void)ws_size;
}
